// TemporalCrossAttention_51376398794965
// MI455X (gfx1250) — matrix operand order repaired and checked by reading, not compiled, not run on hardware
//
#include <hip/hip_runtime.h>
#include <hip/hip_bf16.h>

typedef __attribute__((ext_vector_type(16))) __bf16 v16bf;
typedef __attribute__((ext_vector_type(8)))  __bf16 v8bf;
typedef __attribute__((ext_vector_type(8)))  float  v8f;

static constexpr int Bn = 4, Nn = 4096, Mn = 4096, Cn = 256;

#define WMMA_BF16(a, b, c) \
  __builtin_amdgcn_wmma_f32_16x16x32_bf16(false, (a), false, (b), (short)0, (c), false, false)

// ---- fragment load helpers -------------------------------------------------

// 16 bf16 from two contiguous 8-element runs (A-fragment halves, or B from LDS)
__device__ __forceinline__ v16bf ldbf16_2(const __bf16* p0, const __bf16* p1) {
  v8bf a = *(const v8bf*)p0;
  v8bf b = *(const v8bf*)p1;
  v16bf r;
#pragma unroll
  for (int i = 0; i < 8; ++i) { r[i] = a[i]; r[8 + i] = b[i]; }
  return r;
}
// 16 contiguous bf16 (B-fragment from a row-major operand)
__device__ __forceinline__ v16bf ldbf16_16(const __bf16* p) { return ldbf16_2(p, p + 8); }

// 16 bf16 converted from two contiguous 8-float runs
__device__ __forceinline__ v16bf cvt16_2(const float* p0, const float* p1) {
  v16bf r;
#pragma unroll
  for (int i = 0; i < 8; ++i) { r[i] = (__bf16)p0[i]; r[8 + i] = (__bf16)p1[i]; }
  return r;
}
// 16 bf16 converted from 16 contiguous floats
__device__ __forceinline__ v16bf cvt16_16(const float* p) { return cvt16_2(p, p + 8); }

// ---- kernel 1: linear projection  y = (x @ w^T + b) * oscale  --------------
// x: [R=16384, 256] f32 ; w: [256,256] f32 (row = out feature) ; out bf16
// transposed==0 : out[row][o]  row-major [R][C]
// transposed==1 : out[b][o][m] (V stored as [B][C][M] for contiguous PV B-frags)
// oscale folds the attention 1/sqrt(C) into Q at projection time.
__global__ __launch_bounds__(256) void proj_kernel(
    const float* __restrict__ x, const float* __restrict__ w,
    const float* __restrict__ bias, __bf16* __restrict__ out, int transposed,
    float oscale) {
  const int lane = threadIdx.x & 31;
  const int wv   = threadIdx.x >> 5;
  const int row0 = blockIdx.x * 16;          // 1024 row tiles
  const int m    = lane & 15;
  const int hi   = lane >> 4;
  const int ct0  = wv * 2;                   // 8 waves cover 16 col tiles

  const float* xrow = x + (size_t)(row0 + m) * Cn;
  v8f acc0 = {}, acc1 = {};
#pragma unroll
  for (int c = 0; c < 8; ++c) {
    const int k0 = c * 32 + (hi ? 16 : 0);
    v16bf a = cvt16_2(xrow + k0, xrow + k0 + 8);      // A: 16x32 bf16
    const int i0 = c * 32 + (hi ? 16 : 0);
    v16bf b0 = cvt16_16(w + (size_t)(ct0 * 16 + m) * Cn + i0);        // B = w^T tile
    v16bf b1 = cvt16_16(w + (size_t)((ct0 + 1) * 16 + m) * Cn + i0);
    acc0 = WMMA_BF16(a, b0, acc0);
    acc1 = WMMA_BF16(a, b1, acc1);
  }
  const float bb0 = bias[ct0 * 16 + m];
  const float bb1 = bias[(ct0 + 1) * 16 + m];
#pragma unroll
  for (int r = 0; r < 8; ++r) {
    const int row = row0 + r + hi * 8;
    const float y0 = (acc0[r] + bb0) * oscale;
    const float y1 = (acc1[r] + bb1) * oscale;
    if (!transposed) {
      out[(size_t)row * Cn + ct0 * 16 + m]       = (__bf16)y0;
      out[(size_t)row * Cn + (ct0 + 1) * 16 + m] = (__bf16)y1;
    } else {
      const int b = row / Nn, nl = row % Nn;
      out[((size_t)b * Cn + ct0 * 16 + m) * Mn + nl]       = (__bf16)y0;
      out[((size_t)b * Cn + (ct0 + 1) * 16 + m) * Mn + nl] = (__bf16)y1;
    }
  }
}

// ---- kernel 2: flash attention + matched kpts + flow/confidence ------------
__global__ __launch_bounds__(256) void attn_kernel(
    const __bf16* __restrict__ Qb, const __bf16* __restrict__ Kb,
    const __bf16* __restrict__ Vt, const float* __restrict__ kpred,
    const float* __restrict__ kact, const int* __restrict__ iter_idx,
    __bf16* __restrict__ matched, float* __restrict__ flow,
    float* __restrict__ conf) {
  __shared__ __align__(16) __bf16 pbuf[8][16 * 32];    // per-wave P staging

  const int lane = threadIdx.x & 31;
  const int wv   = threadIdx.x >> 5;
  const int row0 = (blockIdx.x * 8 + wv) * 16;         // global query-row tile
  const int b    = row0 / Nn;
  const int m    = lane & 15;
  const int hi   = lane >> 4;

  const int it = iter_idx[0];
  const float win = (it == 0) ? 64.0f : fmaxf(4.0f, 32.0f * exp2f((float)(1 - it)));

  // rl/kx/ky are PER-LANE partial sums; the row-uniform rescale alpha keeps
  // them consistent, and they are butterfly-reduced once in the epilogue.
  float px[8], py[8], rm[8], rl[8], kx[8], ky[8];
#pragma unroll
  for (int r = 0; r < 8; ++r) {
    const int row = row0 + r + hi * 8;
    px[r] = kpred[(size_t)row * 2];
    py[r] = kpred[(size_t)row * 2 + 1];
    rm[r] = -3.0e38f; rl[r] = 0.f; kx[r] = 0.f; ky[r] = 0.f;
  }
  v8f o[16];
#pragma unroll
  for (int s = 0; s < 16; ++s) o[s] = {};

  const __bf16* Qrow = Qb + (size_t)(row0 + m) * Cn;
  const size_t krow0 = (size_t)b * Mn;

  for (int j = 0; j < Mn; j += 32) {
    // ---- S = Q K^T (Q pre-scaled by 1/sqrt(C)), two 16x16 tiles, K-depth 256
    v8f s0 = {}, s1 = {};
#pragma unroll
    for (int c = 0; c < 8; ++c) {
      const int k0 = c * 32 + (hi ? 16 : 0);
      v16bf aq = ldbf16_2(Qrow + k0, Qrow + k0 + 8);
      const int i0 = c * 32 + (hi ? 16 : 0);
      v16bf bk0 = ldbf16_16(Kb + (krow0 + j + m) * (size_t)Cn + i0);
      v16bf bk1 = ldbf16_16(Kb + (krow0 + j + 16 + m) * (size_t)Cn + i0);
      s0 = WMMA_BF16(aq, bk0, s0);
      s1 = WMMA_BF16(aq, bk1, s1);
    }
    // ---- window mask: masked positions become exactly -1e9 (as reference)
    const float ax0 = kact[(krow0 + j + m) * 2],      ay0 = kact[(krow0 + j + m) * 2 + 1];
    const float ax1 = kact[(krow0 + j + 16 + m) * 2], ay1 = kact[(krow0 + j + 16 + m) * 2 + 1];
#pragma unroll
    for (int r = 0; r < 8; ++r) {
      const bool m0 = (fabsf(px[r] - ax0) < win) && (fabsf(py[r] - ay0) < win);
      const bool m1 = (fabsf(px[r] - ax1) < win) && (fabsf(py[r] - ay1) < win);
      s0[r] = m0 ? s0[r] : -1.0e9f;
      s1[r] = m1 ? s1[r] : -1.0e9f;
    }
    // ---- online softmax: only the row-max needs a per-step butterfly
    float alpha[8]; int resc = 0;
#pragma unroll
    for (int r = 0; r < 8; ++r) {
      float t = fmaxf(s0[r], s1[r]);
      t = fmaxf(t, __shfl_xor(t, 1, 32)); t = fmaxf(t, __shfl_xor(t, 2, 32));
      t = fmaxf(t, __shfl_xor(t, 4, 32)); t = fmaxf(t, __shfl_xor(t, 8, 32));
      const float mn = fmaxf(rm[r], t);
      alpha[r] = __expf(rm[r] - mn);
      resc |= (alpha[r] < 1.0f);
      rm[r] = mn;
      const float p0 = __expf(s0[r] - mn), p1 = __expf(s1[r] - mn);
      s0[r] = p0; s1[r] = p1;
      rl[r] = rl[r] * alpha[r] + (p0 + p1);             // per-lane partials
      kx[r] = kx[r] * alpha[r] + (p0 * ax0 + p1 * ax1);
      ky[r] = ky[r] * alpha[r] + (p0 * ay0 + p1 * ay1);
    }
    if (__ballot(resc)) {                  // rescale O only when a row max moved
      v8f av;
#pragma unroll
      for (int r = 0; r < 8; ++r) av[r] = alpha[r];
#pragma unroll
      for (int s = 0; s < 16; ++s) o[s] *= av;
    }
    // ---- repack P (D-layout) -> A-layout through per-wave LDS slice
    __bf16* pb = pbuf[wv];
#pragma unroll
    for (int r = 0; r < 8; ++r) {
      const int row = r + hi * 8;
      pb[row * 32 + m]      = (__bf16)s0[r];
      pb[row * 32 + 16 + m] = (__bf16)s1[r];
    }
    asm volatile("s_wait_dscnt 0" ::: "memory");   // same-wave cross-lane hand-off
    const int pk0 = hi ? 16 : 0;
    v16bf ap = ldbf16_2(pb + m * 32 + pk0, pb + m * 32 + pk0 + 8);
    // ---- O += P * V  (V transposed -> contiguous B-fragments)
#pragma unroll
    for (int s = 0; s < 16; ++s) {
      const __bf16* vp = Vt + ((size_t)b * Cn + s * 16 + m) * Mn + j + (hi ? 16 : 0);
      v16bf bv = ldbf16_16(vp);
      o[s] = WMMA_BF16(ap, bv, o[s]);
    }
  }

  // ---- epilogue: one-shot butterfly reductions of the deferred partials
#pragma unroll
  for (int r = 0; r < 8; ++r) {
    float ts = rl[r];
    ts += __shfl_xor(ts, 1, 32); ts += __shfl_xor(ts, 2, 32);
    ts += __shfl_xor(ts, 4, 32); ts += __shfl_xor(ts, 8, 32);
    rl[r] = ts;
    float tx = kx[r];
    tx += __shfl_xor(tx, 1, 32); tx += __shfl_xor(tx, 2, 32);
    tx += __shfl_xor(tx, 4, 32); tx += __shfl_xor(tx, 8, 32);
    kx[r] = tx;
    float ty = ky[r];
    ty += __shfl_xor(ty, 1, 32); ty += __shfl_xor(ty, 2, 32);
    ty += __shfl_xor(ty, 4, 32); ty += __shfl_xor(ty, 8, 32);
    ky[r] = ty;
  }
  float inv[8];
#pragma unroll
  for (int r = 0; r < 8; ++r) inv[r] = 1.0f / rl[r];
#pragma unroll
  for (int s = 0; s < 16; ++s)
#pragma unroll
    for (int r = 0; r < 8; ++r) {
      const int row = row0 + r + hi * 8;
      matched[(size_t)row * Cn + s * 16 + m] = (__bf16)(o[s][r] * inv[r]);
    }
  if (m == 0) {
#pragma unroll
    for (int r = 0; r < 8; ++r) {
      const int row = row0 + r + hi * 8;
      const float cv = (rm[r] > -1.0e8f) ? 1.0f : 0.0f;
      flow[(size_t)row * 2]     = (kx[r] * inv[r] - px[r]) * cv;
      flow[(size_t)row * 2 + 1] = (ky[r] * inv[r] - py[r]) * cv;
      conf[row] = cv;
    }
  }
}

// ---- kernel 3: merge GEMM  out = matched @ merge_w^T + b  (f32 out) --------
__global__ __launch_bounds__(256) void merge_kernel(
    const __bf16* __restrict__ xm, const float* __restrict__ w,
    const float* __restrict__ bias, float* __restrict__ out) {
  const int lane = threadIdx.x & 31;
  const int wv   = threadIdx.x >> 5;
  const int row0 = blockIdx.x * 16;
  const int m    = lane & 15;
  const int hi   = lane >> 4;
  const int ct0  = wv * 2;

  const __bf16* xrow = xm + (size_t)(row0 + m) * Cn;
  v8f acc0 = {}, acc1 = {};
#pragma unroll
  for (int c = 0; c < 8; ++c) {
    const int k0 = c * 32 + (hi ? 16 : 0);
    v16bf a = ldbf16_2(xrow + k0, xrow + k0 + 8);
    const int i0 = c * 32 + (hi ? 16 : 0);
    v16bf b0 = cvt16_16(w + (size_t)(ct0 * 16 + m) * Cn + i0);
    v16bf b1 = cvt16_16(w + (size_t)((ct0 + 1) * 16 + m) * Cn + i0);
    acc0 = WMMA_BF16(a, b0, acc0);
    acc1 = WMMA_BF16(a, b1, acc1);
  }
  const float bb0 = bias[ct0 * 16 + m];
  const float bb1 = bias[(ct0 + 1) * 16 + m];
#pragma unroll
  for (int r = 0; r < 8; ++r) {
    const int row = row0 + r + hi * 8;
    out[(size_t)row * Cn + ct0 * 16 + m]       = acc0[r] + bb0;
    out[(size_t)row * Cn + (ct0 + 1) * 16 + m] = acc1[r] + bb1;
  }
}

// ---- kernel 4: tiny geo MLP (3 -> 32 -> silu -> 32) ------------------------
__global__ void geo_kernel(const float* __restrict__ flow,
                           const float* __restrict__ conf,
                           const float* __restrict__ w1, const float* __restrict__ b1,
                           const float* __restrict__ w2, const float* __restrict__ b2,
                           float* __restrict__ out) {
  const int row = blockIdx.x * blockDim.x + threadIdx.x;
  if (row >= Bn * Nn) return;
  const float in0 = flow[(size_t)row * 2], in1 = flow[(size_t)row * 2 + 1], in2 = conf[row];
  float h[32];
#pragma unroll
  for (int o = 0; o < 32; ++o) {
    const float z = w1[o * 3] * in0 + w1[o * 3 + 1] * in1 + w1[o * 3 + 2] * in2 + b1[o];
    h[o] = z / (1.0f + __expf(-z));       // silu
  }
#pragma unroll 4
  for (int o = 0; o < 32; ++o) {
    float z = b2[o];
#pragma unroll
    for (int jj = 0; jj < 32; ++jj) z += w2[o * 32 + jj] * h[jj];
    out[(size_t)row * 32 + o] = z;
  }
}

extern "C" void kernel_launch(void* const* d_in, const int* in_sizes, int n_in,
                              void* d_out, int out_size, void* d_ws, size_t ws_size,
                              hipStream_t stream) {
  const float* nodes_t  = (const float*)d_in[0];
  const float* nodes_t1 = (const float*)d_in[1];
  const float* kpred    = (const float*)d_in[2];
  const float* kact     = (const float*)d_in[3];
  const int*   iter     = (const int*)d_in[4];
  const float* q_w = (const float*)d_in[5];  const float* q_b = (const float*)d_in[6];
  const float* k_w = (const float*)d_in[7];  const float* k_b = (const float*)d_in[8];
  const float* v_w = (const float*)d_in[9];  const float* v_b = (const float*)d_in[10];
  const float* mg_w = (const float*)d_in[11]; const float* mg_b = (const float*)d_in[12];
  const float* g_w1 = (const float*)d_in[13]; const float* g_b1 = (const float*)d_in[14];
  const float* g_w2 = (const float*)d_in[15]; const float* g_b2 = (const float*)d_in[16];

  __bf16* Qb = (__bf16*)d_ws;                       // [B*N, C]   8 MB (pre-scaled)
  __bf16* Kb = Qb + (size_t)Bn * Nn * Cn;           // [B*M, C]   8 MB
  __bf16* Vt = Kb + (size_t)Bn * Mn * Cn;           // [B, C, M]  8 MB
  __bf16* Mf = Vt + (size_t)Bn * Mn * Cn;           // [B*N, C]   8 MB

  float* out_merge = (float*)d_out;                                 // [B*N, C]
  float* out_geo   = out_merge + (size_t)Bn * Nn * Cn;              // [B*N, 32]
  float* out_flow  = out_geo + (size_t)Bn * Nn * 32;                // [B*N, 2]
  float* out_conf  = out_flow + (size_t)Bn * Nn * 2;                // [B*N]

  const dim3 blk(256);
  proj_kernel<<<(Bn * Nn) / 16, blk, 0, stream>>>(nodes_t,  q_w, q_b, Qb, 0, 0.0625f);
  proj_kernel<<<(Bn * Mn) / 16, blk, 0, stream>>>(nodes_t1, k_w, k_b, Kb, 0, 1.0f);
  proj_kernel<<<(Bn * Mn) / 16, blk, 0, stream>>>(nodes_t1, v_w, v_b, Vt, 1, 1.0f);
  attn_kernel<<<(Bn * Nn) / (16 * 8), blk, 0, stream>>>(Qb, Kb, Vt, kpred, kact, iter,
                                                        Mf, out_flow, out_conf);
  merge_kernel<<<(Bn * Nn) / 16, blk, 0, stream>>>(Mf, mg_w, mg_b, out_merge);
  geo_kernel<<<(Bn * Nn + 255) / 256, blk, 0, stream>>>(out_flow, out_conf,
                                                        g_w1, g_b1, g_w2, g_b2, out_geo);
}
